// _SingleDirectionGLA_56118042689800
// MI455X (gfx1250) — matrix operand order repaired and checked by reading, not compiled, not run on hardware
//
#include <hip/hip_runtime.h>
#include <hip/hip_bf16.h>

// ---------------------------------------------------------------------------
// Gated linear attention (single direction), CDNA5 / gfx1250.
//   B=2, L=1024, D=1024, H=8, KH=64, VH=128, M = B*L = 2048, K = 1024.
// Pipeline:
//   1) x: f32->bf16 ; weights: f32->bf16 transposed (N-major) so both WMMA
//      operand fragments are contiguous 16B runs in LDS.
//   2) WMMA bf16 GEMMs: 128x64 block tile, 32x32 per wave (4 wmma/K-step),
//      double-buffered LDS staged by the Tensor Data Mover
//      (tensor_load_to_lds + s_wait_tensorcnt), falling back to
//      global_load_async_to_lds_b128 if the TDM builtin is unavailable.
//   3) low-rank gate, register-resident sequential scan, LN*silu, final GEMM.
// ---------------------------------------------------------------------------

typedef __bf16 v16bf __attribute__((ext_vector_type(16)));
typedef float  v8f   __attribute__((ext_vector_type(8)));
typedef unsigned int u32x4 __attribute__((ext_vector_type(4)));
typedef int          i32x4 __attribute__((ext_vector_type(4)));
typedef int          i32x8 __attribute__((ext_vector_type(8)));

#define MROWS 2048
#define DMODEL 1024
#define DHALF 512
#define NHEADS 8
#define KH 64
#define VH 128
#define LSEQ 1024
#define GK 1024          // shared K of all big GEMMs

#if defined(__has_builtin)
#if __has_builtin(__builtin_amdgcn_tensor_load_to_lds)
#define HAVE_TDM 1
#endif
#endif
#ifndef HAVE_TDM
#define HAVE_TDM 0
#endif

union fragU {
  v16bf v;
  uint4 u[2];
};

__device__ __forceinline__ unsigned lds_off(const void* p) {
  return (unsigned)(uintptr_t)(__attribute__((address_space(3))) const void*)p;
}

#define TM 128
#define TN 64
#define TK 32
#define TKP (TK + 8)   // LDS row stride 80B (TDM pad: 4 DWORDs per 16 DWORDs)

#if HAVE_TDM
// Issue one TDM 2D tile load: tile_rows x TK bf16 elements, row stride GK,
// into LDS at lds_addr with 16B padding per 64B row (matches TKP).
__device__ __forceinline__ void tdm_load_2d(unsigned lds_addr, const void* gptr,
                                            int tile_rows, int tensor_rows) {
  const unsigned long long ga = (unsigned long long)(uintptr_t)gptr;
  u32x4 g0;
  g0[0] = 1u;                                               // count=1, user D#
  g0[1] = lds_addr;                                         // lds_addr
  g0[2] = (unsigned)ga;                                     // global_addr lo
  g0[3] = (unsigned)((ga >> 32) & 0x01FFFFFFu) | 0x80000000u; // ga hi | type=2
  i32x8 g1;
  g1[0] = (1 << 16)        // data_size = 2B
        | (1 << 20)        // pad_enable
        | (3 << 22)        // pad_interval: 16 DWORDs (one 64B tile row)
        | (3 << 25);       // pad_amount: 4 DWORDs (16B)
  g1[1] = (GK & 0xFFFF) << 16;            // tensor_dim0 = K
  g1[2] = (tensor_rows & 0xFFFF) << 16;   // tensor_dim1
  g1[3] = (TK << 16);                     // tile_dim0 = 32
  g1[4] = tile_rows & 0xFFFF;             // tile_dim1
  g1[5] = GK;                             // tensor_dim0_stride = K
  g1[6] = 0;
  g1[7] = 0;
  const i32x4 z4 = {0, 0, 0, 0};
#if __clang_major__ >= 23
  const i32x8 z8 = {0, 0, 0, 0, 0, 0, 0, 0};
  __builtin_amdgcn_tensor_load_to_lds(g0, g1, z4, z4, z8, 0);
#else
  __builtin_amdgcn_tensor_load_to_lds(g0, g1, z4, z4, 0);
#endif
}
#endif

// ---------------------------- f32 -> bf16 ----------------------------------
__global__ __launch_bounds__(256) void k_cvt_bf16(const float* __restrict__ in,
                                                  __bf16* __restrict__ out, int n) {
  int i = blockIdx.x * 256 + threadIdx.x;
  if (i < n) out[i] = (__bf16)in[i];
}

// f32 (K x N) -> bf16 transposed (N x K), tiled through LDS (coalesced both ways)
__global__ __launch_bounds__(256)
void k_cvtT_bf16(const float* __restrict__ in, __bf16* __restrict__ out,
                 int K, int N) {
  __shared__ float tile[32][33];
  const int kb = blockIdx.x * 32, nb = blockIdx.y * 32;
  const int tx = threadIdx.x & 31, ty = threadIdx.x >> 5;  // 32 x 8
#pragma unroll
  for (int i = 0; i < 32; i += 8)
    tile[ty + i][tx] = in[(size_t)(kb + ty + i) * N + nb + tx];
  __syncthreads();
#pragma unroll
  for (int i = 0; i < 32; i += 8)
    out[(size_t)(nb + ty + i) * K + kb + tx] = (__bf16)tile[tx][ty + i];
}

__global__ __launch_bounds__(256) void k_zero_f32(float* __restrict__ p, int n) {
  int i = blockIdx.x * 256 + threadIdx.x;
  if (i < n) p[i] = 0.0f;
}

// ------------------------- WMMA bf16 GEMM ----------------------------------
// C(MxN,f32) = A(MxGK,bf16,row-major) * Bt(NxGK,bf16 == B^T) + bias(n)
// Double-buffered LDS; DMA of tile t+1 overlaps WMMA of tile t.
__global__ __launch_bounds__(256)
void k_gemm_bf16_wmma(const __bf16* __restrict__ A, const __bf16* __restrict__ Bt,
                      float* __restrict__ C, const float* __restrict__ bias,
                      int M, int N) {
  __shared__ __bf16 As[2][TM][TKP];   // 2 x 128 x 40
  __shared__ __bf16 Bs[2][TN][TKP];   // 2 x  64 x 40

  const int t    = threadIdx.x;
  const int lane = t & 31;
  const int wave = t >> 5;
  const int wm   = (wave & 3) * 32;   // wave m-origin
  const int wn   = (wave >> 2) * 32;  // wave n-origin
  const int bm   = blockIdx.y * TM;
  const int bn   = blockIdx.x * TN;

  // fragment coords (constant across K-steps)
  const int lh   = (lane >> 4) * 8;
  const int m0   = wm + (lane & 15);
  const int n0   = wn + (lane & 15);
  const int kgrp = (lane >> 4) * 16;

  v8f acc00 = {}, acc01 = {}, acc10 = {}, acc11 = {};

#if HAVE_TDM
  const bool issuer = (wave == 0);    // one TDM issue per workgroup tile
  const __bf16* gA = A + (size_t)bm * GK;
  const __bf16* gB = Bt + (size_t)bn * GK;
  const unsigned ldsA0 = lds_off(&As[0][0][0]), ldsA1 = lds_off(&As[1][0][0]);
  const unsigned ldsB0 = lds_off(&Bs[0][0][0]), ldsB1 = lds_off(&Bs[1][0][0]);
  if (issuer) {
    tdm_load_2d(ldsA0, gA, TM, M);
    tdm_load_2d(ldsB0, gB, TN, N);
  }
#else
  // per-lane async fallback: each thread stages 3 x 16B per K-step
  const int srow = t >> 2;            // 0..63
  const int scol = (t & 3) * 8;
  const __bf16* gA0 = A + (size_t)(bm + srow) * GK + scol;
  const __bf16* gA1 = A + (size_t)(bm + srow + 64) * GK + scol;
  const __bf16* gB0 = Bt + (size_t)(bn + srow) * GK + scol;
  const unsigned lA0[2] = {lds_off(&As[0][srow][scol]), lds_off(&As[1][srow][scol])};
  const unsigned lA1[2] = {lds_off(&As[0][srow + 64][scol]),
                           lds_off(&As[1][srow + 64][scol])};
  const unsigned lB0[2] = {lds_off(&Bs[0][srow][scol]), lds_off(&Bs[1][srow][scol])};
#define STAGE_ASYNC(bufi, kofs)                                                 \
  do {                                                                          \
    asm volatile("global_load_async_to_lds_b128 %0, %1, off" ::"v"(lA0[bufi]),  \
                 "v"((unsigned long long)(uintptr_t)(gA0 + (kofs))) : "memory");\
    asm volatile("global_load_async_to_lds_b128 %0, %1, off" ::"v"(lA1[bufi]),  \
                 "v"((unsigned long long)(uintptr_t)(gA1 + (kofs))) : "memory");\
    asm volatile("global_load_async_to_lds_b128 %0, %1, off" ::"v"(lB0[bufi]),  \
                 "v"((unsigned long long)(uintptr_t)(gB0 + (kofs))) : "memory");\
  } while (0)
  STAGE_ASYNC(0, 0);
#endif

  int buf = 0;
  for (int kt = 0; kt < GK; kt += TK, buf ^= 1) {
#if HAVE_TDM
    if (issuer) __builtin_amdgcn_s_wait_tensorcnt(0);   // tile kt landed
    __syncthreads();
    if (issuer && kt + TK < GK) {                       // overlap next DMA
      tdm_load_2d(buf ? ldsA0 : ldsA1, gA + kt + TK, TM, M);
      tdm_load_2d(buf ? ldsB0 : ldsB1, gB + kt + TK, TN, N);
    }
#else
    asm volatile("s_wait_asynccnt 0x0" ::: "memory");
    __syncthreads();
    if (kt + TK < GK) STAGE_ASYNC(buf ^ 1, kt + TK);
#endif

    const __bf16(*Ab)[TKP] = As[buf];
    const __bf16(*Bb)[TKP] = Bs[buf];
    // A frag: elements 0..7 = K[lh..lh+7], 8..15 = K[16+lh..16+lh+7]
    fragU a0, a1, b0, b1;
    a0.u[0] = *(const uint4*)&Ab[m0][lh];
    a0.u[1] = *(const uint4*)&Ab[m0][16 + lh];
    a1.u[0] = *(const uint4*)&Ab[m0 + 16][lh];
    a1.u[1] = *(const uint4*)&Ab[m0 + 16][16 + lh];
    // B frag: n = lane%16, elements 0..7 = K[lh..lh+7], 8..15 = K[16+lh..16+lh+7] (Bt is N-major)
    b0.u[0] = *(const uint4*)&Bb[n0][lh];
    b0.u[1] = *(const uint4*)&Bb[n0][16 + lh];
    b1.u[0] = *(const uint4*)&Bb[n0 + 16][lh];
    b1.u[1] = *(const uint4*)&Bb[n0 + 16][16 + lh];

    acc00 = __builtin_amdgcn_wmma_f32_16x16x32_bf16(false, a0.v, false, b0.v,
                                                    (short)0, acc00, false, false);
    acc01 = __builtin_amdgcn_wmma_f32_16x16x32_bf16(false, a0.v, false, b1.v,
                                                    (short)0, acc01, false, false);
    acc10 = __builtin_amdgcn_wmma_f32_16x16x32_bf16(false, a1.v, false, b0.v,
                                                    (short)0, acc10, false, false);
    acc11 = __builtin_amdgcn_wmma_f32_16x16x32_bf16(false, a1.v, false, b1.v,
                                                    (short)0, acc11, false, false);
    __syncthreads();   // protect buf from being overwritten next iteration
  }

  // C layout: VGPR r -> M = r + (lane>=16 ? 8 : 0); N = lane%16
  const int cn0 = bn + wn + (lane & 15);
  const int cm0 = bm + wm + ((lane >> 4) * 8);
  const float bv0 = bias[cn0];
  const float bv1 = bias[cn0 + 16];
  int idx0 = cm0 * N + cn0;           // 32-bit addressing (M*N <= 2^21)
  int idx1 = (cm0 + 16) * N + cn0;
#pragma unroll
  for (int r = 0; r < 8; ++r) {
    C[idx0]      = acc00[r] + bv0;
    C[idx0 + 16] = acc01[r] + bv1;
    C[idx1]      = acc10[r] + bv0;
    C[idx1 + 16] = acc11[r] + bv1;
    idx0 += N;
    idx1 += N;
  }
}

// ----------------------- low-rank gate path --------------------------------
__global__ __launch_bounds__(256) void k_proj_u(const float* __restrict__ x,
                                                const float* __restrict__ Wg1,
                                                float* __restrict__ u) {
  int idx = blockIdx.x * 256 + threadIdx.x;   // 2048*16
  int row = idx >> 4, i = idx & 15;
  float s = 0.f;
  for (int c = 0; c < DMODEL; ++c) s += x[(size_t)row * DMODEL + c] * Wg1[c * 16 + i];
  u[idx] = s;
}

__global__ __launch_bounds__(64) void k_gate(const float* __restrict__ u,
                                             const float* __restrict__ Wg2,
                                             float* __restrict__ gate) {
  int row = blockIdx.x >> 3, h = blockIdx.x & 7, d = threadIdx.x;
  float kg = 0.f;
#pragma unroll
  for (int i = 0; i < 16; ++i)
    kg += u[row * 16 + i] * Wg2[i * DHALF + h * KH + d];
  float ls = fminf(kg, 0.f) - log1pf(__expf(-fabsf(kg)));  // logsigmoid
  __shared__ float red[64];
  red[d] = ls;
  __syncthreads();
  for (int s = 32; s > 0; s >>= 1) {
    if (d < s) red[d] += red[d + s];
    __syncthreads();
  }
  if (d == 0) gate[row * NHEADS + h] = __expf(red[0] * (1.0f / (64.0f * 16.0f)));
}

// ----------------------------- GLA scan ------------------------------------
__global__ __launch_bounds__(128)
void k_gla_scan(const float* __restrict__ q, const float* __restrict__ k,
                const float* __restrict__ v, const float* __restrict__ gate,
                float* __restrict__ att) {
  const int b = blockIdx.x >> 3, h = blockIdx.x & 7;
  const int e = threadIdx.x;
  float s[KH];
#pragma unroll
  for (int d = 0; d < KH; ++d) s[d] = 0.f;

  __shared__ float kq[2 * KH];   // [0..63]=k*scale, [64..127]=q

  for (int l = 0; l < LSEQ; ++l) {
    const int row = b * LSEQ + l;
    if (e < KH)
      kq[e] = k[(size_t)row * DHALF + h * KH + e] * 0.125f;     // 64^-0.5
    else
      kq[e] = q[(size_t)row * DHALF + h * KH + (e - KH)];
    __syncthreads();

    const float gt = gate[row * NHEADS + h] + 1e-6f;
    const float vv = v[(size_t)row * DMODEL + h * VH + e];
    float o = 0.f;
#pragma unroll
    for (int d = 0; d < KH; ++d) {
      s[d] = s[d] * gt + kq[d] * vv;
      o += kq[KH + d] * s[d];
    }
    att[(size_t)row * DMODEL + h * VH + e] = o;
    __syncthreads();
  }
}

// ------------------- LayerNorm(VH) * silu(gp) -> bf16 ----------------------
__global__ __launch_bounds__(128)
void k_ln_gate(const float* __restrict__ att, const float* __restrict__ gp,
               __bf16* __restrict__ y) {
  const int row = blockIdx.x >> 3, h = blockIdx.x & 7, e = threadIdx.x;
  const size_t idx = (size_t)row * DMODEL + h * VH + e;
  const float val = att[idx];
  __shared__ float red[VH];
  red[e] = val;
  __syncthreads();
  for (int s = 64; s > 0; s >>= 1) {
    if (e < s) red[e] += red[e + s];
    __syncthreads();
  }
  const float mu = red[0] * (1.0f / VH);
  __syncthreads();
  const float dv = val - mu;
  red[e] = dv * dv;
  __syncthreads();
  for (int s = 64; s > 0; s >>= 1) {
    if (e < s) red[e] += red[e + s];
    __syncthreads();
  }
  const float var = red[0] * (1.0f / VH);
  const float normed = dv * rsqrtf(var + 1e-5f);
  const float gv = gp[idx];                       // bias already added
  const float g  = gv / (1.0f + __expf(-gv));     // silu
  y[idx] = (__bf16)(g * normed);
}

// ---------------------------------------------------------------------------
extern "C" void kernel_launch(void* const* d_in, const int* in_sizes, int n_in,
                              void* d_out, int out_size, void* d_ws, size_t ws_size,
                              hipStream_t stream) {
  const float* x    = (const float*)d_in[0];
  const float* Wq   = (const float*)d_in[1];
  const float* Wk   = (const float*)d_in[2];
  const float* Wg1  = (const float*)d_in[3];
  const float* Wg2  = (const float*)d_in[4];
  const float* Wv   = (const float*)d_in[5];
  const float* Wgp  = (const float*)d_in[6];
  const float* bgp  = (const float*)d_in[7];
  const float* Wout = (const float*)d_in[8];

  char* ws = (char*)d_ws;
  size_t off = 0;
  auto alloc = [&](size_t bytes) -> void* {
    void* p = ws + off;
    off = (off + bytes + 255) & ~(size_t)255;
    return p;
  };

  __bf16* x_bf    = (__bf16*)alloc((size_t)MROWS * DMODEL * 2);
  __bf16* wqT_bf  = (__bf16*)alloc((size_t)DMODEL * DHALF * 2);   // (512 x 1024)
  __bf16* wkT_bf  = (__bf16*)alloc((size_t)DMODEL * DHALF * 2);
  __bf16* wvT_bf  = (__bf16*)alloc((size_t)DMODEL * DMODEL * 2);  // (1024 x 1024)
  __bf16* wgpT_bf = (__bf16*)alloc((size_t)DMODEL * DMODEL * 2);
  __bf16* woT_bf  = (__bf16*)alloc((size_t)DMODEL * DMODEL * 2);
  float*  q_f     = (float*)alloc((size_t)MROWS * DHALF * 4);
  float*  k_f     = (float*)alloc((size_t)MROWS * DHALF * 4);
  float*  v_f     = (float*)alloc((size_t)MROWS * DMODEL * 4);
  float*  gp_f    = (float*)alloc((size_t)MROWS * DMODEL * 4);
  float*  u_f     = (float*)alloc((size_t)MROWS * 16 * 4);
  float*  gate_f  = (float*)alloc((size_t)MROWS * NHEADS * 4);
  float*  att_f   = (float*)alloc((size_t)MROWS * DMODEL * 4);
  __bf16* y_bf    = (__bf16*)alloc((size_t)MROWS * DMODEL * 2);
  float*  zbias   = (float*)alloc((size_t)DMODEL * 4);

  k_cvt_bf16<<<(MROWS * DMODEL + 255) / 256, 256, 0, stream>>>(x, x_bf,
                                                               MROWS * DMODEL);
  dim3 gT_h(DMODEL / 32, DHALF / 32);
  dim3 gT_f(DMODEL / 32, DMODEL / 32);
  k_cvtT_bf16<<<gT_h, 256, 0, stream>>>(Wq, wqT_bf, DMODEL, DHALF);
  k_cvtT_bf16<<<gT_h, 256, 0, stream>>>(Wk, wkT_bf, DMODEL, DHALF);
  k_cvtT_bf16<<<gT_f, 256, 0, stream>>>(Wv, wvT_bf, DMODEL, DMODEL);
  k_cvtT_bf16<<<gT_f, 256, 0, stream>>>(Wgp, wgpT_bf, DMODEL, DMODEL);
  k_cvtT_bf16<<<gT_f, 256, 0, stream>>>(Wout, woT_bf, DMODEL, DMODEL);
  k_zero_f32<<<(DMODEL + 255) / 256, 256, 0, stream>>>(zbias, DMODEL);

  dim3 gHalf(DHALF / TN, MROWS / TM);    // (8, 16)
  dim3 gFull(DMODEL / TN, MROWS / TM);   // (16, 16)
  k_gemm_bf16_wmma<<<gHalf, 256, 0, stream>>>(x_bf, wqT_bf, q_f, zbias,
                                              MROWS, DHALF);
  k_gemm_bf16_wmma<<<gHalf, 256, 0, stream>>>(x_bf, wkT_bf, k_f, zbias,
                                              MROWS, DHALF);
  k_gemm_bf16_wmma<<<gFull, 256, 0, stream>>>(x_bf, wvT_bf, v_f, zbias,
                                              MROWS, DMODEL);
  k_gemm_bf16_wmma<<<gFull, 256, 0, stream>>>(x_bf, wgpT_bf, gp_f, bgp,
                                              MROWS, DMODEL);

  k_proj_u<<<(MROWS * 16) / 256, 256, 0, stream>>>(x, Wg1, u_f);
  k_gate<<<MROWS * NHEADS, 64, 0, stream>>>(u_f, Wg2, gate_f);

  k_gla_scan<<<2 * NHEADS, 128, 0, stream>>>(q_f, k_f, v_f, gate_f, att_f);
  k_ln_gate<<<MROWS * NHEADS, 128, 0, stream>>>(att_f, gp_f, y_bf);

  k_gemm_bf16_wmma<<<gFull, 256, 0, stream>>>(y_bf, woT_bf, (float*)d_out,
                                              zbias, MROWS, DMODEL);
}
